// ProposalRolloutBranch_13486197309999
// MI455X (gfx1250) — matrix operand order repaired and checked by reading, compiled, not run on hardware
//
#include <hip/hip_runtime.h>
#include <hip/hip_bf16.h>
#include <math.h>

// ---------------------------------------------------------------------------
// ProposalRolloutBranch for MI455X (gfx1250, wave32, WMMA).
// Memory-bound GEMV chain. The 16x-reused step weights are converted to bf16
// ONCE into d_ws, pre-swizzled into the V_WMMA_F32_16X16X32_BF16 B-fragment
// layout; every step GEMV then streams 1KB fragments through WMMA (f32 acc).
// Inner loop is double-buffered so each wave keeps 2 fragments in flight;
// independent GEMV pairs (gi/gh, seed/cond) are fused into single launches.
// Workspace requirement: ~336 MB (assumed available in d_ws).
// ---------------------------------------------------------------------------

typedef __attribute__((ext_vector_type(16))) __bf16 v16bf;
typedef __attribute__((ext_vector_type(8)))  __bf16 v8bf;
typedef __attribute__((ext_vector_type(8)))  float  v8f;

#define DM    4096
#define STEPS 16

__device__ __forceinline__ float gelu_exact(float x) {
  return 0.5f * x * (1.0f + erff(x * 0.70710678118654752f));
}
__device__ __forceinline__ float sigmoidf(float x) {
  return 1.0f / (1.0f + expf(-x));
}

// ---------------------------------------------------------------------------
// fusion = concat(a, p, c, p-a, p*a)   [5*D]
// ---------------------------------------------------------------------------
__global__ void k_fusion(const float* __restrict__ a, const float* __restrict__ p,
                         const float* __restrict__ c, float* __restrict__ fusion) {
  int i = blockIdx.x * 256 + threadIdx.x;
  if (i < DM) {
    float av = a[i], pv = p[i];
    fusion[i]          = av;
    fusion[DM + i]     = pv;
    fusion[2 * DM + i] = c[i];
    fusion[3 * DM + i] = pv - av;
    fusion[4 * DM + i] = pv * av;
  }
}

// ---------------------------------------------------------------------------
// f32 -> bf16 conversion into WMMA B-fragment layout.
// Fragment (jt, kt): 512 bf16 = 32 lanes x 16 values.
// Lane L: col n = L&15, kHalf = L>>4; element i holds W[kt*32+kHalf*16+i][jt*16+n].
// Fragment order: frag = jt * nK + kt  (K-tiles contiguous within a j-tile).
// trans==0: src(k,j) = src[k*J + j];  trans==1: src(k,j) = src[j*K + k].
// ---------------------------------------------------------------------------
__global__ void k_convert(const float* __restrict__ src, __bf16* __restrict__ dst,
                          int K, int J, int trans) {
  int nK = K >> 5;
  int nJ = J >> 4;
  long long total = (long long)nK * nJ * 32;
  long long tid = (long long)blockIdx.x * 256 + threadIdx.x;
  if (tid >= total) return;
  int L    = (int)(tid & 31);
  int frag = (int)(tid >> 5);
  int kt = frag % nK;
  int jt = frag / nK;
  int n    = L & 15;
  int half = L >> 4;
  int j     = jt * 16 + n;
  int kbase = kt * 32 + half * 16;

  union { __bf16 e[16]; v16bf v; } u;
  if (trans) {
    const float4* s4 = (const float4*)(src + (size_t)j * K + kbase);
    float4 f0 = s4[0], f1 = s4[1], f2 = s4[2], f3 = s4[3];
    const float fs[16] = {f0.x, f0.y, f0.z, f0.w, f1.x, f1.y, f1.z, f1.w,
                          f2.x, f2.y, f2.z, f2.w, f3.x, f3.y, f3.z, f3.w};
#pragma unroll
    for (int i = 0; i < 16; ++i) u.e[i] = (__bf16)fs[i];
  } else {
#pragma unroll
    for (int i = 0; i < 16; ++i)
      u.e[i] = (__bf16)src[(size_t)(kbase + i) * J + j];
  }
  *(v16bf*)(dst + ((size_t)frag << 9) + ((size_t)L << 4)) = u.v;
}

// ---------------------------------------------------------------------------
// Plain f32 GEMV pair (for once-used seed/cond weights), two independent
// GEMVs fused into one launch via blockIdx.y. Block: 512 threads =
// 64 cols x 8 K-slices. mode: 0=GELU, 1=linear, 2=resid + 0.1*val.
// ---------------------------------------------------------------------------
__global__ void k_gemv_f32_pair(const float* __restrict__ W0, const float* __restrict__ W1,
                                const float* __restrict__ b0, const float* __restrict__ b1,
                                const float* __restrict__ x0, const float* __restrict__ x1,
                                float* __restrict__ o0, float* __restrict__ o1,
                                int K, int J, int mode0, int mode1,
                                const float* __restrict__ resid) {
  __shared__ float red[512];
  int set = blockIdx.y;
  const float* W    = set ? W1 : W0;
  const float* bias = set ? b1 : b0;
  const float* x    = set ? x1 : x0;
  float*       out  = set ? o1 : o0;
  int          mode = set ? mode1 : mode0;

  int t  = threadIdx.x;
  int jj = t & 63;
  int ks = t >> 6;                       // 0..7
  int j  = blockIdx.x * 64 + jj;
  float acc = 0.0f;
#pragma unroll 4
  for (int k = ks; k < K; k += 8)
    acc += x[k] * W[(size_t)k * J + j];
  red[t] = acc;
  __syncthreads();
  if (ks == 0) {
    float s = bias[j];
#pragma unroll
    for (int w = 0; w < 8; ++w) s += red[jj + w * 64];
    if (mode == 0)      s = gelu_exact(s);
    else if (mode == 2) s = resid[j] + 0.1f * s;
    out[j] = s;
  }
}

// ---------------------------------------------------------------------------
// WMMA bf16 GEMV (optionally two independent GEMVs fused: blocks with
// blockIdx.x >= nJt1 run the second parameter set; K/act shared).
// x (f32) is staged to LDS as bf16, replicated across all 16 A rows (per ISA
// 16-bit A layout), so every lane's c[0] is a valid output.
// Block: 512 threads = 16 waves sharing one 16-output j-tile, splitting K.
// Inner loop is double-buffered: next A/B fragments are loaded before the
// current WMMA so each wave keeps 2 fragments in flight.
// If x==nullptr (set 0 only), gather x = concat(condition, state, stepv).
// ---------------------------------------------------------------------------
__global__ void k_wmma_gemv(const __bf16* __restrict__ Wfrag,
                            const float* __restrict__ bias,
                            const float* __restrict__ x,
                            float* __restrict__ out,
                            const __bf16* __restrict__ Wfrag2,
                            const float* __restrict__ bias2,
                            const float* __restrict__ x2,
                            float* __restrict__ out2,
                            const float* __restrict__ condition,
                            const float* __restrict__ state,
                            const float* __restrict__ stepv,
                            int nJt1, int K, int actGelu) {
  __shared__ __align__(16) __bf16 xsh[3 * DM];
  __shared__ float red[16][16];

  int set = (blockIdx.x >= nJt1) ? 1 : 0;
  const __bf16* Wf = set ? Wfrag2 : Wfrag;
  const float*  bi = set ? bias2  : bias;
  const float*  xe = set ? x2     : x;
  float*        ou = set ? out2   : out;
  int jt = set ? (blockIdx.x - nJt1) : blockIdx.x;

  int t = threadIdx.x;
  for (int k = t; k < K; k += 512) {
    float v;
    if (xe) v = xe[k];
    else    v = (k < DM) ? condition[k]
              : (k < 2 * DM) ? state[k - DM] : stepv[k - 2 * DM];
    xsh[k] = (__bf16)v;
  }
  __syncthreads();

  int wave  = t >> 5;
  int lane  = t & 31;
  int nK    = K >> 5;
  int nIter = nK >> 4;                   // K-tiles per wave (K mult. of 512)
  int half8 = (lane >> 4) << 4;          // 0 for lanes 0-15, 16 for lanes 16-31

  const v8bf*   xs8  = (const v8bf*)xsh;
  const __bf16* wbase = Wf + ((size_t)jt * nK << 9) + (lane << 4);

  v8f c = {0.f, 0.f, 0.f, 0.f, 0.f, 0.f, 0.f, 0.f};

  // prologue: load first fragment pair
  int k0 = (wave << 5);
  union { v8bf h[2]; v16bf v; } ua;
  ua.h[0] = xs8[(k0 + half8) >> 3];
  ua.h[1] = xs8[(k0 + 8 + half8) >> 3];
  v16bf a0 = ua.v;
  v16bf b0 = *(const v16bf*)(wbase + ((size_t)wave << 9));

  for (int i = 1; i < nIter; ++i) {      // uniform trip count: EXEC all-1s
    int kt = wave + (i << 4);
    int kk = kt << 5;
    union { v8bf h[2]; v16bf v; } un;
    un.h[0] = xs8[(kk + half8) >> 3];
    un.h[1] = xs8[(kk + 8 + half8) >> 3];
    v16bf a1 = un.v;
    v16bf b1 = *(const v16bf*)(wbase + ((size_t)kt << 9));
    c = __builtin_amdgcn_wmma_f32_16x16x32_bf16(false, a0, false, b0,
                                                (short)0, c, false, false);
    a0 = a1;
    b0 = b1;
  }
  c = __builtin_amdgcn_wmma_f32_16x16x32_bf16(false, a0, false, b0,
                                              (short)0, c, false, false);

  if (lane < 16) red[wave][lane] = c[0];
  __syncthreads();
  if (t < 16) {
    float s = bi[jt * 16 + t];
#pragma unroll
    for (int w = 0; w < 16; ++w) s += red[w][t];
    if (actGelu) s = gelu_exact(s);
    ou[jt * 16 + t] = s;
  }
}

// ---------------------------------------------------------------------------
// GRU pointwise update + LayerNorm. One block of 1024 threads, 4 elems each.
// state <- h (raw);   outStates <- LN(h).
// ---------------------------------------------------------------------------
__global__ void k_gru_ln(const float* __restrict__ gi, const float* __restrict__ gh,
                         float* __restrict__ state,
                         const float* __restrict__ lng, const float* __restrict__ lnb,
                         float* __restrict__ outStates) {
  __shared__ float s1[1024];
  __shared__ float s2[1024];
  int t = threadIdx.x;
  float h[4];
  float lsum = 0.f, lsq = 0.f;
#pragma unroll
  for (int i = 0; i < 4; ++i) {
    int j = t + i * 1024;
    float ir = gi[j], iz = gi[DM + j], inn = gi[2 * DM + j];
    float hr = gh[j], hz = gh[DM + j], hn  = gh[2 * DM + j];
    float r = sigmoidf(ir + hr);
    float z = sigmoidf(iz + hz);
    float n = tanhf(inn + r * hn);
    float s = state[j];
    float hv = (1.0f - z) * n + z * s;
    h[i] = hv;
    state[j] = hv;
    lsum += hv;
    lsq  += hv * hv;
  }
  s1[t] = lsum; s2[t] = lsq;
  __syncthreads();
  for (int s = 512; s > 0; s >>= 1) {
    if (t < s) { s1[t] += s1[t + s]; s2[t] += s2[t + s]; }
    __syncthreads();
  }
  float mu   = s1[0] * (1.0f / DM);
  float var  = s2[0] * (1.0f / DM) - mu * mu;
  float rstd = rsqrtf(var + 1e-5f);
#pragma unroll
  for (int i = 0; i < 4; ++i) {
    int j = t + i * 1024;
    outStates[j] = (h[i] - mu) * rstd * lng[j] + lnb[j];
  }
}

// ---------------------------------------------------------------------------
// gate[t] = sigmoid(dot(states[t], gw[0:D]) + dot(condition, gw[D:2D]) + gb)
// ---------------------------------------------------------------------------
__global__ void k_gate(const float* __restrict__ states, const float* __restrict__ condition,
                       const float* __restrict__ gw, const float* __restrict__ gb,
                       float* __restrict__ gate) {
  __shared__ float red[256];
  int t = threadIdx.x;
  int step = blockIdx.x;
  float acc = 0.f;
  for (int j = t; j < DM; j += 256)
    acc += states[step * DM + j] * gw[j] + condition[j] * gw[DM + j];
  red[t] = acc;
  __syncthreads();
  for (int s = 128; s > 0; s >>= 1) {
    if (t < s) red[t] += red[t + s];
    __syncthreads();
  }
  if (t == 0) gate[step] = sigmoidf(red[0] + gb[0]);
}

__global__ void k_summary(const float* __restrict__ states, const float* __restrict__ gate,
                          float* __restrict__ summary) {
  int j = blockIdx.x * 256 + threadIdx.x;
  if (j >= DM) return;
  float gs = 0.f, num = 0.f;
#pragma unroll
  for (int t = 0; t < STEPS; ++t) {
    float g = gate[t];
    gs  += g;
    num += g * states[t * DM + j];
  }
  summary[j] = num / fmaxf(gs, 1e-6f);
}

// ---------------------------------------------------------------------------
extern "C" void kernel_launch(void* const* d_in, const int* in_sizes, int n_in,
                              void* d_out, int out_size, void* d_ws, size_t ws_size,
                              hipStream_t stream) {
  const float* anchor   = (const float*)d_in[0];
  const float* proposal = (const float*)d_in[1];
  const float* context  = (const float*)d_in[2];
  const float* seed_w1  = (const float*)d_in[3];
  const float* seed_b1  = (const float*)d_in[4];
  const float* seed_w2  = (const float*)d_in[5];
  const float* seed_b2  = (const float*)d_in[6];
  const float* cond_w1  = (const float*)d_in[7];
  const float* cond_b1  = (const float*)d_in[8];
  const float* cond_w2  = (const float*)d_in[9];
  const float* cond_b2  = (const float*)d_in[10];
  const float* in_w1    = (const float*)d_in[11];
  const float* in_b1    = (const float*)d_in[12];
  const float* in_w2    = (const float*)d_in[13];
  const float* in_b2    = (const float*)d_in[14];
  const float* step_emb = (const float*)d_in[15];
  const float* w_ih     = (const float*)d_in[16];
  const float* b_ih     = (const float*)d_in[17];
  const float* w_hh     = (const float*)d_in[18];
  const float* b_hh     = (const float*)d_in[19];
  const float* ln_g     = (const float*)d_in[20];
  const float* ln_b     = (const float*)d_in[21];
  const float* gate_w   = (const float*)d_in[22];
  const float* gate_b   = (const float*)d_in[23];

  float* states  = (float*)d_out;                 // [16, 4096]
  float* summary = (float*)d_out + STEPS * DM;    // [4096]

  // ---- workspace layout -------------------------------------------------
  char*  ws  = (char*)d_ws;
  size_t off = 0;
  auto take = [&](size_t bytes) -> void* {
    void* p = ws + off;
    off = (off + bytes + 255) & ~(size_t)255;
    return p;
  };
  __bf16* in_w1_f = (__bf16*)take((size_t)3 * DM * DM * 2);  // K=12288,J=4096
  __bf16* in_w2_f = (__bf16*)take((size_t)DM * DM * 2);      // K=4096, J=4096
  __bf16* w_ih_f  = (__bf16*)take((size_t)DM * 3 * DM * 2);  // K=4096, J=12288 (transposed)
  __bf16* w_hh_f  = (__bf16*)take((size_t)DM * 3 * DM * 2);
  float* fusion    = (float*)take((size_t)5 * DM * 4);
  float* h1_seed   = (float*)take(DM * 4);
  float* h1_cond   = (float*)take(DM * 4);
  float* condition = (float*)take(DM * 4);
  float* state     = (float*)take(DM * 4);
  float* h1        = (float*)take(DM * 4);
  float* xout      = (float*)take(DM * 4);
  float* gi        = (float*)take((size_t)3 * DM * 4);
  float* gh        = (float*)take((size_t)3 * DM * 4);
  float* gateBuf   = (float*)take(STEPS * 4);
  (void)ws_size; (void)in_sizes; (void)n_in; (void)out_size;

  // ---- fusion vector + bf16 weight conversion (fragment layout) ---------
  k_fusion<<<(DM + 255) / 256, 256, 0, stream>>>(anchor, proposal, context, fusion);

  {
    long long th;
    th = (long long)(3 * DM / 32) * (DM / 16) * 32;       // in_w1
    k_convert<<<(unsigned)((th + 255) / 256), 256, 0, stream>>>(in_w1, in_w1_f, 3 * DM, DM, 0);
    th = (long long)(DM / 32) * (DM / 16) * 32;           // in_w2
    k_convert<<<(unsigned)((th + 255) / 256), 256, 0, stream>>>(in_w2, in_w2_f, DM, DM, 0);
    th = (long long)(DM / 32) * (3 * DM / 16) * 32;       // w_ih^T, w_hh^T
    k_convert<<<(unsigned)((th + 255) / 256), 256, 0, stream>>>(w_ih, w_ih_f, DM, 3 * DM, 1);
    k_convert<<<(unsigned)((th + 255) / 256), 256, 0, stream>>>(w_hh, w_hh_f, DM, 3 * DM, 1);
  }

  // ---- seed/cond MLPs (f32, weights used once), pairs fused -------------
  {
    dim3 g1(DM / 64, 2);
    // stage 1: h1_seed / h1_cond = GELU(fusion @ {seed,cond}_w1 + b1)
    k_gemv_f32_pair<<<g1, 512, 0, stream>>>(seed_w1, cond_w1, seed_b1, cond_b1,
                                            fusion, fusion, h1_seed, h1_cond,
                                            5 * DM, DM, 0, 0, nullptr);
    // stage 2: condition = h1_cond @ cond_w2 + b2 ; state = p + 0.1*(h1_seed @ seed_w2 + b2)
    k_gemv_f32_pair<<<g1, 512, 0, stream>>>(cond_w2, seed_w2, cond_b2, seed_b2,
                                            h1_cond, h1_seed, condition, state,
                                            DM, DM, 1, 2, proposal);
  }

  // ---- 16 sequential rollout steps (bf16 WMMA GEMVs) --------------------
  for (int t = 0; t < STEPS; ++t) {
    const float* stepv = step_emb + (size_t)t * DM;
    // h1 = GELU(concat(cond,state,step) @ in_w1 + in_b1)   J=4096, K=12288
    k_wmma_gemv<<<DM / 16, 512, 0, stream>>>(in_w1_f, in_b1, nullptr, h1,
                                             in_w1_f, in_b1, nullptr, h1,
                                             condition, state, stepv,
                                             DM / 16, 3 * DM, 1);
    // xout = h1 @ in_w2 + in_b2                            J=4096, K=4096
    k_wmma_gemv<<<DM / 16, 512, 0, stream>>>(in_w2_f, in_b2, h1, xout,
                                             in_w2_f, in_b2, h1, xout,
                                             nullptr, nullptr, nullptr,
                                             DM / 16, DM, 0);
    // gi = xout @ w_ih^T + b_ih ; gh = state @ w_hh^T + b_hh  (fused pair)
    k_wmma_gemv<<<2 * (3 * DM / 16), 512, 0, stream>>>(w_ih_f, b_ih, xout, gi,
                                                       w_hh_f, b_hh, state, gh,
                                                       nullptr, nullptr, nullptr,
                                                       3 * DM / 16, DM, 0);
    // GRU pointwise + LayerNorm
    k_gru_ln<<<1, 1024, 0, stream>>>(gi, gh, state, ln_g, ln_b, states + (size_t)t * DM);
  }

  // ---- gated summary ----------------------------------------------------
  k_gate<<<STEPS, 256, 0, stream>>>(states, condition, gate_w, gate_b, gateBuf);
  k_summary<<<DM / 256, 256, 0, stream>>>(states, gateBuf, summary);
}
